// GraphTransformerLayer_70866960384543
// MI455X (gfx1250) — matrix operand order repaired and checked by reading, not compiled, not run on hardware
//
#include <hip/hip_runtime.h>

// ---------------------------------------------------------------------------
// Types for CDNA5 WMMA (wave32, v_wmma_f32_16x16x32_bf16)
// ---------------------------------------------------------------------------
typedef __attribute__((ext_vector_type(16))) __bf16 bf16x16;
typedef __attribute__((ext_vector_type(8)))  __bf16 bf16x8;
typedef __attribute__((ext_vector_type(4)))  __bf16 bf16x4;
typedef __attribute__((ext_vector_type(8)))  float  f32x8;

#define LDA 136  // padded LDS row stride in bf16 elems (272B -> bank-conflict-free)

__device__ __forceinline__ bf16x16 frag_cat(bf16x8 lo, bf16x8 hi) {
    return __builtin_shufflevector(lo, hi, 0,1,2,3,4,5,6,7,8,9,10,11,12,13,14,15);
}

__device__ __forceinline__ float wredsum(float v) {
    #pragma unroll
    for (int o = 16; o > 0; o >>= 1) v += __shfl_xor(v, o, 32);
    return v;
}

// order-preserving float <-> uint encoding for atomicMax on floats
__device__ __forceinline__ unsigned fenc(float f) {
    unsigned u = __float_as_uint(f);
    return (u & 0x80000000u) ? ~u : (u | 0x80000000u);
}
__device__ __forceinline__ float fdec(unsigned u) {
    unsigned b = (u & 0x80000000u) ? (u & 0x7FFFFFFFu) : ~u;
    return __uint_as_float(b);
}

// ---------------------------------------------------------------------------
// GEMM: O[M x Ncols] = act(A[M x K] @ W[Ncols x K]^T + bias)
// bf16 operands staged in LDS, fp32 accumulate via v_wmma_f32_16x16x32_bf16.
// Block = 256 threads (8 waves). Block tile = 128 rows x 128 cols.
// Each wave: 16-row band x 128 cols = 8 WMMA accumulator tiles.
// Per K-step: batch-load A frag + all 8 B frags, then a back-to-back WMMA
// clause (independent accumulators -> no hazard NOPs, single dscnt wait).
// Requires: K % 128 == 0, Ncols % 128 == 0 (true for all call sites).
// ---------------------------------------------------------------------------
__global__ __launch_bounds__(256) void gemm_bf16_kernel(
    const float* __restrict__ A, const float* __restrict__ W,
    const float* __restrict__ bias, float* __restrict__ O,
    int M, int K, int Ncols, int relu)
{
    __shared__ __bf16 sA[128 * LDA];
    __shared__ __bf16 sB[128 * LDA];

    const int rowbase = blockIdx.x * 128;
    const int colbase = blockIdx.y * 128;
    const int t    = threadIdx.x;
    const int w    = t >> 5;        // wave id 0..7
    const int lane = t & 31;
    const int mr   = lane & 15;     // row/col within 16x16 tile
    const int h    = lane >> 4;     // lane half

    f32x8 acc[8];
    #pragma unroll
    for (int i = 0; i < 8; ++i)
        #pragma unroll
        for (int r = 0; r < 8; ++r) acc[i][r] = 0.f;

    const int sr    = t >> 1;        // staging row 0..127 (2 threads/row)
    const int halfk = (t & 1) * 64;  // 64 floats per thread per row

    for (int kb = 0; kb < K; kb += 128) {
        // ---- stage A tile (rows of activations), fp32 -> bf16 ----
        {
            const int  grow  = rowbase + sr;
            const bool valid = grow < M;
            const float* src = A + (size_t)grow * K + kb + halfk;
            #pragma unroll
            for (int i = 0; i < 16; ++i) {
                float4 f = valid ? *(const float4*)(src + i * 4)
                                 : make_float4(0.f, 0.f, 0.f, 0.f);
                bf16x4 bv;
                bv[0] = (__bf16)f.x; bv[1] = (__bf16)f.y;
                bv[2] = (__bf16)f.z; bv[3] = (__bf16)f.w;
                *(bf16x4*)&sA[sr * LDA + halfk + i * 4] = bv;
            }
        }
        // ---- stage W tile (rows of W = output cols), fp32 -> bf16 ----
        {
            const int gcol   = colbase + sr;
            const float* src = W + (size_t)gcol * K + kb + halfk;
            #pragma unroll
            for (int i = 0; i < 16; ++i) {
                float4 f = *(const float4*)(src + i * 4);
                bf16x4 bv;
                bv[0] = (__bf16)f.x; bv[1] = (__bf16)f.y;
                bv[2] = (__bf16)f.z; bv[3] = (__bf16)f.w;
                *(bf16x4*)&sB[sr * LDA + halfk + i * 4] = bv;
            }
        }
        __syncthreads();

        const __bf16* arow = &sA[(w * 16 + mr) * LDA];
        #pragma unroll
        for (int ks = 0; ks < 128; ks += 32) {
            // A frag (16x32 bf16): lane half h holds K runs [h*8,h*8+8), [16+h*8,...)
            bf16x16 afrag = frag_cat(*(const bf16x8*)(arow + ks + h * 8),
                                     *(const bf16x8*)(arow + ks + 16 + h * 8));
            // Batch-load all 8 B frags (32x16 bf16): lane half h holds K runs
            // [h*8,h*8+8), [16+h*8,...). One ds_load burst -> one wait.
            bf16x16 bfrag[8];
            #pragma unroll
            for (int tt = 0; tt < 8; ++tt) {
                const __bf16* brow = &sB[(tt * 16 + mr) * LDA + ks + h * 8];
                bfrag[tt] = frag_cat(*(const bf16x8*)brow,
                                     *(const bf16x8*)(brow + 16));
            }
            // Back-to-back WMMA clause: independent accumulators, no RAW.
            #pragma unroll
            for (int tt = 0; tt < 8; ++tt) {
                acc[tt] = __builtin_amdgcn_wmma_f32_16x16x32_bf16(
                    false, afrag, false, bfrag[tt], (short)0, acc[tt], false, false);
            }
        }
        __syncthreads();
    }

    // ---- epilogue: bias + optional ReLU, C/D layout M = r + 8*h, N = mr ----
    #pragma unroll
    for (int tt = 0; tt < 8; ++tt) {
        const int col = colbase + tt * 16 + mr;
        const float bv = bias ? bias[col] : 0.f;
        #pragma unroll
        for (int r = 0; r < 8; ++r) {
            const int grow = rowbase + w * 16 + h * 8 + r;
            if (grow < M) {
                float v = acc[tt][r] + bv;
                if (relu) v = fmaxf(v, 0.f);
                O[(size_t)grow * Ncols + col] = v;
            }
        }
    }
}

// ---------------------------------------------------------------------------
// Edge pass 1: alpha[e,h] = dot(q[dst], k[src] + e_feat)/sqrt(C); segment max.
// One wave per edge: lane owns 4 contiguous dims (head = lane>>2).
// e_feat recomputed from edge_attr @ We^T with We cached in LDS (6KB).
// ---------------------------------------------------------------------------
__global__ __launch_bounds__(256) void edge_alpha_kernel(
    const int* __restrict__ src, const int* __restrict__ dst,
    const float* __restrict__ ea, const float* __restrict__ We,
    const float* __restrict__ q, const float* __restrict__ k,
    float* __restrict__ alpha, unsigned* __restrict__ amax, int E)
{
    __shared__ float sWe[128 * 12];
    for (int i = threadIdx.x; i < 128 * 12; i += 256) sWe[i] = We[i];
    __syncthreads();

    const int e    = blockIdx.x * 8 + (threadIdx.x >> 5);
    const int lane = threadIdx.x & 31;
    if (e >= E) return;

    const int s  = src[e], d = dst[e];
    const int d0 = lane * 4;

    float4 qv = *(const float4*)(q + (size_t)d * 128 + d0);
    float4 kv = *(const float4*)(k + (size_t)s * 128 + d0);

    float eav = (lane < 12) ? ea[(size_t)e * 12 + lane] : 0.f;
    float ef0 = 0.f, ef1 = 0.f, ef2 = 0.f, ef3 = 0.f;
    #pragma unroll
    for (int j = 0; j < 12; ++j) {
        float aj = __shfl(eav, j, 32);
        ef0 += aj * sWe[(d0 + 0) * 12 + j];
        ef1 += aj * sWe[(d0 + 1) * 12 + j];
        ef2 += aj * sWe[(d0 + 2) * 12 + j];
        ef3 += aj * sWe[(d0 + 3) * 12 + j];
    }

    float p = qv.x * (kv.x + ef0) + qv.y * (kv.y + ef1)
            + qv.z * (kv.z + ef2) + qv.w * (kv.w + ef3);
    p += __shfl_xor(p, 1, 32);   // reduce over the 4 lanes of one head
    p += __shfl_xor(p, 2, 32);
    p *= 0.25f;                  // 1/sqrt(16)

    if ((lane & 3) == 0) {
        const int hh = lane >> 2;
        alpha[(size_t)e * 8 + hh] = p;
        atomicMax(&amax[(size_t)d * 8 + hh], fenc(p));
    }
}

// ---------------------------------------------------------------------------
// Edge pass 2: ex = exp(alpha - max); segment sum of ex.
// ---------------------------------------------------------------------------
__global__ __launch_bounds__(256) void edge_softmax_kernel(
    const int* __restrict__ dst, float* __restrict__ alpha,
    const unsigned* __restrict__ amax, float* __restrict__ denom, int E)
{
    const int tid = blockIdx.x * 256 + threadIdx.x;
    if (tid >= E * 8) return;
    const int e = tid >> 3, hh = tid & 7;
    const int d = dst[e];
    const float m  = fdec(amax[(size_t)d * 8 + hh]);
    const float ex = __expf(alpha[tid] - m);
    alpha[tid] = ex;
    atomicAdd(&denom[(size_t)d * 8 + hh], ex);
}

// ---------------------------------------------------------------------------
// Edge pass 3: outacc[dst] += ex * (v[src] + e_feat).
// ---------------------------------------------------------------------------
__global__ __launch_bounds__(256) void edge_aggregate_kernel(
    const int* __restrict__ src, const int* __restrict__ dst,
    const float* __restrict__ ea, const float* __restrict__ We,
    const float* __restrict__ v, const float* __restrict__ exw,
    float* __restrict__ outacc, int E)
{
    __shared__ float sWe[128 * 12];
    for (int i = threadIdx.x; i < 128 * 12; i += 256) sWe[i] = We[i];
    __syncthreads();

    const int e    = blockIdx.x * 8 + (threadIdx.x >> 5);
    const int lane = threadIdx.x & 31;
    if (e >= E) return;

    const int s  = src[e], d = dst[e];
    const int d0 = lane * 4;

    float4 vv = *(const float4*)(v + (size_t)s * 128 + d0);

    float eav = (lane < 12) ? ea[(size_t)e * 12 + lane] : 0.f;
    float ef0 = 0.f, ef1 = 0.f, ef2 = 0.f, ef3 = 0.f;
    #pragma unroll
    for (int j = 0; j < 12; ++j) {
        float aj = __shfl(eav, j, 32);
        ef0 += aj * sWe[(d0 + 0) * 12 + j];
        ef1 += aj * sWe[(d0 + 1) * 12 + j];
        ef2 += aj * sWe[(d0 + 2) * 12 + j];
        ef3 += aj * sWe[(d0 + 3) * 12 + j];
    }

    const float wgt = exw[(size_t)e * 8 + (lane >> 2)];
    float* base = outacc + (size_t)d * 128 + d0;
    atomicAdd(base + 0, wgt * (vv.x + ef0));
    atomicAdd(base + 1, wgt * (vv.y + ef1));
    atomicAdd(base + 2, wgt * (vv.z + ef2));
    atomicAdd(base + 3, wgt * (vv.w + ef3));
}

// ---------------------------------------------------------------------------
// Node fuse: softmax divide, beta-gated skip, residual + LayerNorm1 -> h1.
// One wave per node.
// ---------------------------------------------------------------------------
__global__ __launch_bounds__(256) void node_fuse_kernel(
    const float* __restrict__ outacc, const float* __restrict__ denom,
    const float* __restrict__ xr, const float* __restrict__ x,
    const float* __restrict__ Wbeta, const float* __restrict__ g1,
    const float* __restrict__ be1, float* __restrict__ h1, int Nn)
{
    const int wid  = (blockIdx.x * 256 + threadIdx.x) >> 5;
    const int lane = threadIdx.x & 31;
    if (wid >= Nn) return;

    const int d0 = lane * 4, head = lane >> 2;
    const float dn = denom[(size_t)wid * 8 + head] + 1e-16f;

    float4 o4 = *(const float4*)(outacc + (size_t)wid * 128 + d0);
    float o[4] = { o4.x / dn, o4.y / dn, o4.z / dn, o4.w / dn };
    float4 r4 = *(const float4*)(xr + (size_t)wid * 128 + d0);
    float r[4] = { r4.x, r4.y, r4.z, r4.w };

    float part = 0.f;
    #pragma unroll
    for (int c = 0; c < 4; ++c)
        part += Wbeta[d0 + c] * o[c] + Wbeta[128 + d0 + c] * r[c]
              + Wbeta[256 + d0 + c] * (o[c] - r[c]);
    const float bdot = wredsum(part);
    const float beta = 1.f / (1.f + __expf(-bdot));

    float4 xv = *(const float4*)(x + (size_t)wid * 128 + d0);
    float tv[4];
    tv[0] = beta * r[0] + (1.f - beta) * o[0] + xv.x;
    tv[1] = beta * r[1] + (1.f - beta) * o[1] + xv.y;
    tv[2] = beta * r[2] + (1.f - beta) * o[2] + xv.z;
    tv[3] = beta * r[3] + (1.f - beta) * o[3] + xv.w;

    const float mu = wredsum(tv[0] + tv[1] + tv[2] + tv[3]) * (1.f / 128.f);
    float vp = 0.f;
    #pragma unroll
    for (int c = 0; c < 4; ++c) { float dlt = tv[c] - mu; vp += dlt * dlt; }
    const float var = wredsum(vp) * (1.f / 128.f);
    const float inv = rsqrtf(var + 1e-5f);

    float* dst = h1 + (size_t)wid * 128 + d0;
    #pragma unroll
    for (int c = 0; c < 4; ++c)
        dst[c] = (tv[c] - mu) * inv * g1[d0 + c] + be1[d0 + c];
}

// ---------------------------------------------------------------------------
// Final: out = LayerNorm(f2 + h1) * g2 + be2. One wave per node.
// ---------------------------------------------------------------------------
__global__ __launch_bounds__(256) void final_ln_kernel(
    const float* __restrict__ f2, const float* __restrict__ h1,
    const float* __restrict__ g2, const float* __restrict__ be2,
    float* __restrict__ out, int Nn)
{
    const int wid  = (blockIdx.x * 256 + threadIdx.x) >> 5;
    const int lane = threadIdx.x & 31;
    if (wid >= Nn) return;

    const int d0 = lane * 4;
    float4 a = *(const float4*)(f2 + (size_t)wid * 128 + d0);
    float4 c = *(const float4*)(h1 + (size_t)wid * 128 + d0);
    float tv[4] = { a.x + c.x, a.y + c.y, a.z + c.z, a.w + c.w };

    const float mu = wredsum(tv[0] + tv[1] + tv[2] + tv[3]) * (1.f / 128.f);
    float vp = 0.f;
    #pragma unroll
    for (int i = 0; i < 4; ++i) { float dlt = tv[i] - mu; vp += dlt * dlt; }
    const float var = wredsum(vp) * (1.f / 128.f);
    const float inv = rsqrtf(var + 1e-5f);

    float* dst = out + (size_t)wid * 128 + d0;
    #pragma unroll
    for (int i = 0; i < 4; ++i)
        dst[i] = (tv[i] - mu) * inv * g2[d0 + i] + be2[d0 + i];
}

// ---------------------------------------------------------------------------
extern "C" void kernel_launch(void* const* d_in, const int* in_sizes, int n_in,
                              void* d_out, int out_size, void* d_ws, size_t ws_size,
                              hipStream_t stream) {
    (void)n_in; (void)out_size; (void)ws_size;

    const float* x     = (const float*)d_in[0];
    const int*   ei    = (const int*)  d_in[1];   // [2,E]: row0=src, row1=dst
    const float* ea    = (const float*)d_in[2];
    const float* Wq    = (const float*)d_in[3];
    const float* bq    = (const float*)d_in[4];
    const float* Wk    = (const float*)d_in[5];
    const float* bk    = (const float*)d_in[6];
    const float* Wv    = (const float*)d_in[7];
    const float* bv    = (const float*)d_in[8];
    const float* We    = (const float*)d_in[9];
    const float* Wskip = (const float*)d_in[10];
    const float* bskip = (const float*)d_in[11];
    const float* Wbeta = (const float*)d_in[12];
    const float* g1    = (const float*)d_in[13];
    const float* be1   = (const float*)d_in[14];
    const float* g2    = (const float*)d_in[15];
    const float* be2   = (const float*)d_in[16];
    const float* W1    = (const float*)d_in[17];
    const float* b1    = (const float*)d_in[18];
    const float* W2    = (const float*)d_in[19];
    const float* b2    = (const float*)d_in[20];

    const int N = in_sizes[0] / 128;
    const int E = in_sizes[2] / 12;

    float* ws = (float*)d_ws;
    const size_t ND = (size_t)N * 128;

    // Workspace layout (f1 aliases q/k/v/outacc, which are dead before FFN):
    float*    qb     = ws;             // [N,128]
    float*    kbuf   = ws + ND;        // [N,128]
    float*    vbuf   = ws + 2 * ND;    // [N,128]
    float*    outacc = ws + 3 * ND;    // [N,128]
    float*    f1     = ws;             // [N,512]  (aliases the 4 buffers above)
    float*    xrb    = ws + 4 * ND;    // [N,128]
    float*    h1     = ws + 5 * ND;    // [N,128]
    float*    f2     = ws + 6 * ND;    // [N,128]
    float*    alphab = ws + 7 * ND;    // [E,8]
    unsigned* amaxb  = (unsigned*)(alphab + (size_t)E * 8); // [N,8]
    float*    denomb = (float*)(amaxb + (size_t)N * 8);     // [N,8]

    const dim3 blk(256);
    const int  rb = (N + 127) / 128;

    hipMemsetAsync(amaxb,  0, (size_t)N * 8 * 4, stream);   // 0 == encoded bottom
    hipMemsetAsync(denomb, 0, (size_t)N * 8 * 4, stream);
    hipMemsetAsync(outacc, 0, ND * 4, stream);

    // q/k/v/skip projections (WMMA GEMMs)
    gemm_bf16_kernel<<<dim3(rb, 1), blk, 0, stream>>>(x, Wq,    bq,    qb,   N, 128, 128, 0);
    gemm_bf16_kernel<<<dim3(rb, 1), blk, 0, stream>>>(x, Wk,    bk,    kbuf, N, 128, 128, 0);
    gemm_bf16_kernel<<<dim3(rb, 1), blk, 0, stream>>>(x, Wv,    bv,    vbuf, N, 128, 128, 0);
    gemm_bf16_kernel<<<dim3(rb, 1), blk, 0, stream>>>(x, Wskip, bskip, xrb,  N, 128, 128, 0);

    // sparse attention (memory-bound rooflined part)
    edge_alpha_kernel<<<(E + 7) / 8, blk, 0, stream>>>(ei, ei + E, ea, We, qb, kbuf,
                                                       alphab, amaxb, E);
    edge_softmax_kernel<<<(E * 8 + 255) / 256, blk, 0, stream>>>(ei + E, alphab, amaxb,
                                                                 denomb, E);
    edge_aggregate_kernel<<<(E + 7) / 8, blk, 0, stream>>>(ei, ei + E, ea, We, vbuf,
                                                           alphab, outacc, E);

    // beta-gated skip + residual + LN1
    node_fuse_kernel<<<(N + 7) / 8, blk, 0, stream>>>(outacc, denomb, xrb, x, Wbeta,
                                                      g1, be1, h1, N);

    // FFN (WMMA GEMMs) + final LN
    gemm_bf16_kernel<<<dim3(rb, 4), blk, 0, stream>>>(h1, W1, b1, f1, N, 128, 512, 1);
    gemm_bf16_kernel<<<dim3(rb, 1), blk, 0, stream>>>(f1, W2, b2, f2, N, 512, 128, 0);
    final_ln_kernel<<<(N + 7) / 8, blk, 0, stream>>>(f2, h1, g2, be2, (float*)d_out, N);
}
